// WorldDisc_68685116997871
// MI455X (gfx1250) — matrix operand order repaired and checked by reading, not compiled, not run on hardware
//
#include <hip/hip_runtime.h>
#include <hip/hip_bf16.h>
#include <math.h>
#include <type_traits>

// ---------------------------------------------------------------------------
// Types for CDNA5 WMMA (wave32): D(16x16,f32) = A(16x32,f16) * B(32x16,f16) + C
// ---------------------------------------------------------------------------
typedef __attribute__((ext_vector_type(16))) _Float16 v16h;
typedef __attribute__((ext_vector_type(8)))  float    v8f;

union AFrag { v16h v; _Float16 h[16]; };

// compile-time for-loop so k-chunk indices are literals (forces full folding
// of k/TAPS, k%TAPS into immediate LDS offsets)
template<int N, typename F>
__device__ __forceinline__ void static_for(F&& f) {
    if constexpr (N > 0) {
        static_for<N - 1>(static_cast<F&&>(f));
        f(std::integral_constant<int, N - 1>{});
    }
}

__device__ __forceinline__ float sigmoidf_(float x) {
    return 1.0f / (1.0f + __expf(-x));
}

// Problem constants (from the reference)
#define BB   64
#define TT   64
#define CC   3
#define HH   64
#define WW   64
#define HID  8
#define NL   32          /* 4*HID gate channels            */
#define KL   99          /* (C+HID)*9 im2col taps          */
#define KCHL 4           /* ceil(99/32)                    */

// ---------------------------------------------------------------------------
// A-matrix 16x32 f16 fragment build (ISA 7.12.2):
//   lanes 0-15 (hi=0): M=lane,    half j -> k_local = (j<8 ? j : j+8)
//   lanes16-31 (hi=1): M=lane-16, half j -> k_local = (j<8 ? j : j+8) + 8
// Branching on hi makes k a literal inside each arm, so the (ci,ky,kx)
// decomposition folds to immediate offsets: pure ds_load_u16, no div VALU.
// ---------------------------------------------------------------------------
template<int KC, int KTOT, int TAPS, int KW, int KH, int RSTRIDE>
__device__ __forceinline__ v16h build_afrag(const _Float16* __restrict__ rows,
                                            int xbase, int hi)
{
    AFrag a;
    if (hi == 0) {
#pragma unroll
        for (int j = 0; j < 16; ++j) {
            const int k = KC * 32 + (j < 8 ? j : j + 8);
            _Float16 v = (_Float16)0.0f;
            if (k < KTOT) {
                const int ci = k / TAPS, tp = k % TAPS;
                v = rows[(ci * KH + tp / KW) * RSTRIDE + xbase + (tp % KW)];
            }
            a.h[j] = v;
        }
    } else {
#pragma unroll
        for (int j = 0; j < 16; ++j) {
            const int k = KC * 32 + (j < 8 ? j : j + 8) + 8;
            _Float16 v = (_Float16)0.0f;
            if (k < KTOT) {
                const int ci = k / TAPS, tp = k % TAPS;
                v = rows[(ci * KH + tp / KW) * RSTRIDE + xbase + (tp % KW)];
            }
            a.h[j] = v;
        }
    }
    return a.v;
}

// ---------------------------------------------------------------------------
// Weight packing: B-fragment layout.
// B 32x16 f16 layout: lane n=l&15, hi=l>>4; half j -> k_local = (j<8 ? j : j+8) + hi*8.
// wp[((nt*KCH + kc)*32 + lane)*16 + j];  K-order: k = ci*TAPS + tap.
// ---------------------------------------------------------------------------
__global__ void pack_weights_kernel(const float* __restrict__ w,
                                    _Float16* __restrict__ wp,
                                    int COUT, int CIN, int TAPS, int KCH)
{
    int total = (COUT / 16) * KCH * 32 * 16;
    for (int idx = blockIdx.x * blockDim.x + threadIdx.x; idx < total;
         idx += gridDim.x * blockDim.x) {
        int j    = idx & 15;
        int lane = (idx >> 4) & 31;
        int kc   = (idx >> 9) % KCH;
        int nt   = (idx >> 9) / KCH;
        int n    = nt * 16 + (lane & 15);
        int hi   = lane >> 4;
        int k    = kc * 32 + (j < 8 ? j : j + 8) + hi * 8;
        float v  = 0.0f;
        if (k < CIN * TAPS) {
            int ci = k / TAPS, tp = k % TAPS;
            v = w[((size_t)n * CIN + ci) * TAPS + tp];
        }
        wp[idx] = (_Float16)v;
    }
}

__global__ void zero_kernel(float* __restrict__ p, size_t n) {
    size_t i = (size_t)blockIdx.x * blockDim.x + threadIdx.x;
    if (i < n) p[i] = 0.0f;
}

// ---------------------------------------------------------------------------
// ConvLSTM step. One block = one (batch, row). 4 waves cover x=0..63 in
// 16-wide M tiles. GEMM: M=16 positions, N=32 gates (2 WMMA N-tiles),
// K=99 taps padded to 128 (4 chunks of 32).
// ---------------------------------------------------------------------------
__global__ __launch_bounds__(128)
void lstm_step_kernel(const float* __restrict__ x,      // (B,T,C,H,W)
                      const float* __restrict__ h_in,   // (B,HID,H,W)
                      float*       __restrict__ h_out,  // (B,HID,H,W)
                      float*       __restrict__ c_buf,  // (B,HID,H,W)
                      const _Float16* __restrict__ wp,  // packed [2][4][32][16]
                      const float* __restrict__ bias,   // (32,)
                      int t)
{
    __shared__ _Float16 rows[33 * 66];       // (C+HID)*3 halo rows, x in [-1,64]
    __shared__ float    glds[4][NL][16];     // per-wave gate tiles

    const int b   = blockIdx.x / HH;
    const int y   = blockIdx.x % HH;
    const int tid = threadIdx.x;

    // ---- stage x_t and h rows (with zero halo) as f16 ----
    for (int i = tid; i < 33 * 66; i += 128) {
        int r  = i / 66, xp = i % 66;
        int ci = r / 3,  ky = r % 3;
        int yy = y + ky - 1;
        int xx = xp - 1;
        float v = 0.0f;
        if (yy >= 0 && yy < HH && xx >= 0 && xx < WW) {
            if (ci < CC)
                v = x[((((size_t)b * TT + t) * CC + ci) * HH + yy) * WW + xx];
            else
                v = h_in[(((size_t)b * HID + (ci - CC)) * HH + yy) * WW + xx];
        }
        rows[i] = (_Float16)v;
    }
    // prefetch next timestep's input rows (gfx1250 global_prefetch_b8)
    if (tid == 0 && t + 1 < TT) {
        __builtin_prefetch(x + ((((size_t)b * TT + (t + 1)) * CC) * HH + y) * WW, 0, 1);
    }
    __syncthreads();

    const int wid = tid >> 5, lane = tid & 31;
    const int m = lane & 15, hi = lane >> 4;
    const int x0 = wid * 16;
    const int xbase = x0 + m;   // halo shift already built into `rows` columns

    // ---- load all B fragments (2 N-tiles x 4 K-chunks) ----
    v16h bf[2][KCHL];
#pragma unroll
    for (int nt = 0; nt < 2; ++nt)
#pragma unroll
        for (int kc = 0; kc < KCHL; ++kc)
            bf[nt][kc] = *(const v16h*)(wp + (((nt * KCHL + kc) * 32 + lane) << 4));

    v8f acc0 = {}, acc1 = {};
    static_for<KCHL>([&](auto KCc) {
        constexpr int KC = KCc.value;
        v16h a = build_afrag<KC, KL, 9, 3, 3, 66>(rows, xbase, hi);
        acc0 = __builtin_amdgcn_wmma_f32_16x16x32_f16(false, a, false, bf[0][KC],
                                                      (short)0, acc0, false, false);
        acc1 = __builtin_amdgcn_wmma_f32_16x16x32_f16(false, a, false, bf[1][KC],
                                                      (short)0, acc1, false, false);
    });

    // ---- D layout: n = lane&15 (+16), m = r + hi*8; transpose via LDS ----
#pragma unroll
    for (int r = 0; r < 8; ++r) {
        glds[wid][(lane & 15)][r + hi * 8]      = acc0[r] + bias[lane & 15];
        glds[wid][(lane & 15) + 16][r + hi * 8] = acc1[r] + bias[(lane & 15) + 16];
    }
    __syncthreads();

    // ---- fused LSTM cell update: 8 hid channels x 64 x positions ----
    for (int e = tid; e < HID * WW; e += 128) {
        int xq = e & 63;
        int ch = e >> 6;
        int wv = xq >> 4, xl = xq & 15;
        float ig = sigmoidf_(glds[wv][ch][xl]);
        float fg = sigmoidf_(glds[wv][HID + ch][xl]);
        float og = sigmoidf_(glds[wv][2 * HID + ch][xl]);
        float gg = tanhf(glds[wv][3 * HID + ch][xl]);
        size_t idx  = (((size_t)b * HID + ch) * HH + y) * WW + xq;
        float c_old = c_buf[idx];
        float c_new = fg * c_old + ig * gg;
        c_buf[idx]  = c_new;
        h_out[idx]  = og * tanhf(c_new);
    }
}

// ---------------------------------------------------------------------------
// Generic 5x5 VALID conv + bias + ReLU via WMMA.
// One block = one (batch, output row). WX waves tile output x in 16s;
// wave-group g = wid/WX handles N-tiles nt = g, g+G, ... (G = NWAVES/WX).
// A fragment built once per K-chunk and shared across this wave's N-tiles.
// LDS rows are staged with 16 zero-padded columns so no per-element bound
// check is needed in the A build.
// ---------------------------------------------------------------------------
template<int CIN, int COUT, int HIN, int WIN, int NWAVES, int WX>
__global__ __launch_bounds__(NWAVES * 32)
void conv5_wmma_kernel(const float* __restrict__ in,    // (B,CIN,HIN,WIN)
                       const _Float16* __restrict__ wp, // packed fragments
                       const float* __restrict__ bias,  // (COUT,)
                       float* __restrict__ out)         // (B,COUT,OH,OW)
{
    constexpr int OH   = HIN - 4;
    constexpr int OW   = WIN - 4;
    constexpr int KTOT = CIN * 25;
    constexpr int KCH  = (KTOT + 31) / 32;
    constexpr int NT   = COUT / 16;
    constexpr int G    = NWAVES / WX;    // wave-groups along N
    constexpr int NTW  = NT / G;         // N-tiles per wave
    constexpr int RW   = WIN + 16;       // zero-padded row stride

    __shared__ _Float16 rows[CIN * 5 * RW];

    const int b   = blockIdx.x / OH;
    const int oy  = blockIdx.x % OH;
    const int tid = threadIdx.x;

    for (int i = tid; i < CIN * 5 * RW; i += NWAVES * 32) {
        int r  = i / RW, xx = i % RW;
        int ci = r / 5,  ky = r % 5;
        float v = 0.0f;
        if (xx < WIN)
            v = in[(((size_t)b * CIN + ci) * HIN + (oy + ky)) * WIN + xx];
        rows[i] = (_Float16)v;
    }
    __syncthreads();

    const int wid = tid >> 5, lane = tid & 31;
    const int m = lane & 15, hi = lane >> 4;
    const int x0 = (wid % WX) * 16;
    const int xbase = x0 + m;
    const int g = wid / WX;

    v8f acc[NTW];
#pragma unroll
    for (int i = 0; i < NTW; ++i) acc[i] = (v8f){};

    static_for<KCH>([&](auto KCc) {
        constexpr int KC = KCc.value;
        v16h a = build_afrag<KC, KTOT, 25, 5, 5, RW>(rows, xbase, hi);
#pragma unroll
        for (int i = 0; i < NTW; ++i) {
            int nt = g + i * G;
            v16h bfr = *(const v16h*)(wp + (((nt * KCH + KC) * 32 + lane) << 4));
            acc[i] = __builtin_amdgcn_wmma_f32_16x16x32_f16(false, a, false, bfr,
                                                            (short)0, acc[i], false, false);
        }
    });

#pragma unroll
    for (int i = 0; i < NTW; ++i) {
        int nt = g + i * G;
        int n = nt * 16 + (lane & 15);
        float bv = bias[n];
#pragma unroll
        for (int r = 0; r < 8; ++r) {
            int ox = x0 + r + hi * 8;
            if (ox < OW) {
                float v = acc[i][r] + bv;
                out[(((size_t)b * COUT + n) * OH + oy) * OW + ox] = v > 0.0f ? v : 0.0f;
            }
        }
    }
}

// ---------------------------------------------------------------------------
// 3x3 stride-3 VALID max pool (B implicit = 64)
// ---------------------------------------------------------------------------
__global__ void pool3_kernel(const float* __restrict__ in, float* __restrict__ out,
                             int CH, int HIN, int WIN)
{
    int OH = HIN / 3, OW = WIN / 3;
    int total = BB * CH * OH * OW;
    int idx = blockIdx.x * blockDim.x + threadIdx.x;
    if (idx >= total) return;
    int ox = idx % OW; int tmp = idx / OW;
    int oy = tmp % OH; tmp /= OH;
    int ch = tmp % CH; int b = tmp / CH;
    const float* base = in + (((size_t)b * CH + ch) * HIN + oy * 3) * WIN + ox * 3;
    float mx = -INFINITY;
#pragma unroll
    for (int ky = 0; ky < 3; ++ky)
#pragma unroll
        for (int kx = 0; kx < 3; ++kx) {
            float v = base[ky * WIN + kx];
            mx = v > mx ? v : mx;
        }
    out[idx] = mx;
}

// conv3: (B,64,5,5) -> (B,128), 5x5 VALID == full dot, + ReLU
__global__ void conv3_kernel(const float* __restrict__ in,
                             const float* __restrict__ w,   // (128,64,5,5)
                             const float* __restrict__ bias,
                             float* __restrict__ out)       // (B,128)
{
    int idx = blockIdx.x * blockDim.x + threadIdx.x;
    if (idx >= BB * 128) return;
    int b = idx / 128, co = idx % 128;
    const float* ip  = in + (size_t)b * 64 * 25;
    const float* wpp = w + (size_t)co * 64 * 25;
    float s = bias[co];
    for (int i = 0; i < 64 * 25; ++i) s += ip[i] * wpp[i];
    out[idx] = s > 0.0f ? s : 0.0f;
}

// MLP 128->64 (ReLU) then head 64->1
__global__ __launch_bounds__(64)
void mlp_head_kernel(const float* __restrict__ feat,   // (B,128)
                     const float* __restrict__ w_mlp,  // (64,128)
                     const float* __restrict__ b_mlp,  // (64,)
                     const float* __restrict__ w_head, // (1,64)
                     const float* __restrict__ b_head, // (1,)
                     float* __restrict__ out)          // (B,1)
{
    __shared__ float part[64];
    int b = blockIdx.x, j = threadIdx.x;
    const float* f = feat + (size_t)b * 128;
    float s = b_mlp[j];
    for (int k = 0; k < 128; ++k) s += w_mlp[j * 128 + k] * f[k];
    s = s > 0.0f ? s : 0.0f;
    part[j] = s * w_head[j];
    __syncthreads();
    if (j == 0) {
        float tsum = 0.0f;
        for (int k = 0; k < 64; ++k) tsum += part[k];
        out[b] = tsum + b_head[0];
    }
}

// ---------------------------------------------------------------------------
extern "C" void kernel_launch(void* const* d_in, const int* in_sizes, int n_in,
                              void* d_out, int out_size, void* d_ws, size_t ws_size,
                              hipStream_t stream)
{
    (void)in_sizes; (void)n_in; (void)out_size; (void)ws_size;

    const float* x      = (const float*)d_in[0];
    const float* w_lstm = (const float*)d_in[1];
    const float* b_lstm = (const float*)d_in[2];
    const float* w1     = (const float*)d_in[3];
    const float* b1     = (const float*)d_in[4];
    const float* w2     = (const float*)d_in[5];
    const float* b2     = (const float*)d_in[6];
    const float* w3     = (const float*)d_in[7];
    const float* b3     = (const float*)d_in[8];
    const float* w_mlp  = (const float*)d_in[9];
    const float* b_mlp  = (const float*)d_in[10];
    const float* w_head = (const float*)d_in[11];
    const float* b_head = (const float*)d_in[12];
    float* out = (float*)d_out;

    // workspace carve-up (all chunks are multiples of 256 bytes)
    char* ws = (char*)d_ws;
    const size_t NHC = (size_t)BB * HID * HH * WW;          // 2,097,152 elems
    size_t o = 0;
    float* h0   = (float*)(ws + o); o += NHC * 4;
    float* h1   = (float*)(ws + o); o += NHC * 4;
    float* cbuf = (float*)(ws + o); o += NHC * 4;
    float* big  = (float*)(ws + o); o += (size_t)BB * 32 * 60 * 60 * 4; // feat1 / feat2
    float* pool = (float*)(ws + o); o += (size_t)BB * 32 * 20 * 20 * 4; // pool1 / pool2
    float* feat3 = (float*)(ws + o); o += (size_t)BB * 128 * 4;
    _Float16* wpl = (_Float16*)(ws + o); o += (size_t)2 * 4  * 512 * 2; // lstm pack
    _Float16* wp1 = (_Float16*)(ws + o); o += (size_t)2 * 7  * 512 * 2; // conv1 pack
    _Float16* wp2 = (_Float16*)(ws + o); o += (size_t)4 * 25 * 512 * 2; // conv2 pack

    // 1) pack weights into WMMA B-fragment layout
    pack_weights_kernel<<<64, 256, 0, stream>>>(w_lstm, wpl, NL, CC + HID, 9, 4);
    pack_weights_kernel<<<64, 256, 0, stream>>>(w1, wp1, 32, HID, 25, 7);
    pack_weights_kernel<<<64, 256, 0, stream>>>(w2, wp2, 64, 32, 25, 25);

    // 2) zero h0,h1,c (contiguous)
    {
        size_t n = NHC * 3;
        zero_kernel<<<(unsigned)((n + 255) / 256), 256, 0, stream>>>(h0, n);
    }

    // 3) ConvLSTM scan: 64 sequential steps, h ping-pong, c in place
    for (int t = 0; t < TT; ++t) {
        const float* hin  = (t & 1) ? h1 : h0;
        float*       hout = (t & 1) ? h0 : h1;
        lstm_step_kernel<<<BB * HH, 128, 0, stream>>>(x, hin, hout, cbuf,
                                                      wpl, b_lstm, t);
    }
    float* h_last = h0;   // after 64 steps (even count) final h is back in h0

    // 4) head: conv1 (8->32, 64x64 -> 60x60) + ReLU  [WMMA]
    conv5_wmma_kernel<HID, 32, 64, 64, 4, 4><<<BB * 60, 128, 0, stream>>>(
        h_last, wp1, b1, big);

    // 5) pool3: (B,32,60,60) -> (B,32,20,20)
    pool3_kernel<<<(BB * 32 * 20 * 20 + 255) / 256, 256, 0, stream>>>(
        big, pool, 32, 60, 60);

    // 6) conv2 (32->64, 20x20 -> 16x16) + ReLU  [WMMA]  (reuses `big`)
    conv5_wmma_kernel<32, 64, 20, 20, 4, 1><<<BB * 16, 128, 0, stream>>>(
        pool, wp2, b2, big);

    // 7) pool3: (B,64,16,16) -> (B,64,5,5)   (reuses `pool`)
    pool3_kernel<<<(BB * 64 * 5 * 5 + 255) / 256, 256, 0, stream>>>(
        big, pool, 64, 16, 16);

    // 8) conv3 (64->128, 5x5 full) + ReLU -> (B,128)
    conv3_kernel<<<(BB * 128 + 255) / 256, 256, 0, stream>>>(pool, w3, b3, feat3);

    // 9) MLP + head -> (B,1)
    mlp_head_kernel<<<BB, 64, 0, stream>>>(feat3, w_mlp, b_mlp, w_head, b_head, out);
}
